// CRFModel_35278861369606
// MI455X (gfx1250) — compile-verified, matrix operand order checked by reading, not run on hardware
//
#include <hip/hip_runtime.h>
#include <hip/hip_bf16.h>
#include <hip/hip_fp16.h>

typedef __attribute__((ext_vector_type(16))) _Float16 v16h;
typedef __attribute__((ext_vector_type(8)))  _Float16 v8h;
typedef __attribute__((ext_vector_type(8)))  float    v8f;

#define BSZ 64
#define TSEQ 512
#define EDIM 1024
#define HDIM 200
#define GDIM 800      // 4*H
#define HPAD 224      // H padded to multiple of 32 (recurrent K)
#define DSTR 416      // 2*H = 400 padded to multiple of 32 (layer-1 input K / h stride)
#define MROWS 32768   // B*T
#define KTAGS 9

__device__ __forceinline__ v16h cat8h(v8h lo, v8h hi) {
  return __builtin_shufflevector(lo, hi, 0,1,2,3,4,5,6,7,8,9,10,11,12,13,14,15);
}

__device__ __forceinline__ float sigm(float x) { return 1.f / (1.f + __expf(-x)); }

// ---------------------------------------------------------------------------
// Convert f32 (rows x K, row-major) -> f16 (rows x Kpad), zero padding.
// ---------------------------------------------------------------------------
__global__ void convpad_kernel(const float* __restrict__ src, _Float16* __restrict__ dst,
                               int K, int Kpad, int total) {
  int gid = blockIdx.x * blockDim.x + threadIdx.x;
  if (gid >= total) return;
  int row = gid / Kpad, k = gid % Kpad;
  dst[gid] = (k < K) ? (_Float16)src[row * K + k] : (_Float16)0.f;
}

__global__ void zero_kernel(float* __restrict__ p, int n) {
  int gid = blockIdx.x * blockDim.x + threadIdx.x;
  if (gid < n) p[gid] = 0.f;
}

// ---------------------------------------------------------------------------
// XP[M=32768, N=800] = A[M,Kpad](f32, stride Kpad, zero-padded) @
//                      W16[N,Kpad](f16, zero-padded)^T + (b1+b2)[N]
// Block = 320 thr = 10 waves (2 M-tiles x 5 N-tiles); block tile 32(M) x 80(N).
// Grid (1024, 10) covers M,N exactly -> no guards, WMMA at full EXEC.
// ---------------------------------------------------------------------------
__global__ __launch_bounds__(320) void gemm_xp_kernel(
    const float* __restrict__ A, const _Float16* __restrict__ W16, int Kpad,
    const float* __restrict__ b1, const float* __restrict__ b2,
    float* __restrict__ XP)
{
  __shared__ _Float16 As[32][32];
  const int tid  = threadIdx.x;
  const int lane = tid & 31;
  const int wave = tid >> 5;       // 0..9
  const int wm   = wave & 1;       // M sub-tile
  const int wn   = wave >> 1;      // N sub-tile 0..4
  const int mBase = blockIdx.x * 32;
  const int nCol  = blockIdx.y * 80 + wn * 16 + (lane & 15);
  const int r    = lane & 15;
  const int half = lane >> 4;

  // hoisted pointers
  const float*    aRow = A + (size_t)(mBase + (tid >> 3)) * Kpad + ((tid & 7) << 2);
  const _Float16* wPtr = W16 + (size_t)nCol * Kpad + half * 8;
  const _Float16* ap   = &As[wm * 16 + r][half * 8];

  v8f acc = {};
  const int kIters = Kpad >> 5;
  for (int it = 0; it < kIters; ++it) {
    const int kOff = it << 5;
    if (tid < 256) {               // stage 32x32 A tile, f32 -> f16
      float4 v = *(const float4*)(aRow + kOff);
      const int row = tid >> 3, kk = (tid & 7) << 2;
      As[row][kk + 0] = (_Float16)v.x;
      As[row][kk + 1] = (_Float16)v.y;
      As[row][kk + 2] = (_Float16)v.z;
      As[row][kk + 3] = (_Float16)v.w;
      __builtin_prefetch(aRow + kOff + 32, 0, 1);   // next A chunk
    }
    __syncthreads();
    v16h af = cat8h(*(const v8h*)ap, *(const v8h*)(ap + 16));
    const _Float16* bp = wPtr + kOff;
    v16h bf = cat8h(*(const v8h*)bp, *(const v8h*)(bp + 16));
    __builtin_prefetch(bp + 32, 0, 1);              // next W chunk
    acc = __builtin_amdgcn_wmma_f32_16x16x32_f16(false, af, false, bf,
                                                 (short)0, acc, false, false);
    __syncthreads();
  }
  const float bias = b1[nCol] + b2[nCol];
#pragma unroll
  for (int v = 0; v < 8; ++v) {
    int m = mBase + wm * 16 + v + 8 * half;
    XP[(size_t)m * GDIM + nCol] = acc[v] + bias;
  }
}

// ---------------------------------------------------------------------------
// One bidirectional LSTM layer: grid = 2 blocks (dir 0 fwd, dir 1 bwd),
// block = 1024 threads = 32 waves. Sequential scan over T; h(f16) + c(f32) +
// gate buffer G(f32 64x800) all LDS-resident (~285 KB < 320 KB/WGP).
// Per step: G = H(64x224 f16) @ whh16(800x224)^T (WMMA) + xp[:,t,:]; gates.
// hout rows have stride DSTR=416 (pre-zeroed pad cols feed layer-1 GEMM).
// ---------------------------------------------------------------------------
__global__ __launch_bounds__(1024) void lstm_layer_kernel(
    const float* __restrict__ xpf, const float* __restrict__ xpb,
    const _Float16* __restrict__ whhf, const _Float16* __restrict__ whhb,
    float* __restrict__ hout)
{
  extern __shared__ char smem[];
  float*    G  = (float*)smem;                                     // 64*800 f32
  _Float16* Hs = (_Float16*)(smem + BSZ * GDIM * 4);               // 64*224 f16
  float*    Cs = (float*)(smem + BSZ * GDIM * 4 + BSZ * HPAD * 2); // 64*200 f32

  const int dir = blockIdx.x;
  const float* __restrict__ xp = dir ? xpb : xpf;
  const _Float16* __restrict__ whh = dir ? whhb : whhf;
  const int dirOff = dir ? HDIM : 0;

  const int tid  = threadIdx.x;
  const int lane = tid & 31;
  const int wave = tid >> 5;
  const int r    = lane & 15;
  const int half = lane >> 4;

  for (int e = tid; e < BSZ * HDIM; e += 1024) Cs[e] = 0.f;
  for (int e = tid; e < BSZ * HPAD; e += 1024) Hs[e] = (_Float16)0.f;
  __syncthreads();

  for (int ts = 0; ts < TSEQ; ++ts) {
    const int t = dir ? (TSEQ - 1 - ts) : ts;
    // ---- GEMM phase: 4 (M) x 50 (N) = 200 tiles over 32 waves ----
    for (int tile = wave; tile < 200; tile += 32) {
      const int mT = (tile & 3) << 4;
      const int nT = (tile >> 2) << 4;
      const int nCol = nT + r;
      const _Float16* apb = &Hs[(mT + r) * HPAD + half * 8];
      const _Float16* bpb = whh + (size_t)nCol * HPAD + half * 8;
      v8f acc = {};
#pragma unroll
      for (int kb = 0; kb < HPAD; kb += 32) {
        v16h af = cat8h(*(const v8h*)(apb + kb), *(const v8h*)(apb + kb + 16));
        v16h bf = cat8h(*(const v8h*)(bpb + kb), *(const v8h*)(bpb + kb + 16));
        acc = __builtin_amdgcn_wmma_f32_16x16x32_f16(false, af, false, bf,
                                                     (short)0, acc, false, false);
      }
#pragma unroll
      for (int v = 0; v < 8; ++v) {
        int m = mT + v + 8 * half;                 // batch index 0..63
        G[m * GDIM + nCol] = acc[v] + xp[((size_t)m * TSEQ + t) * GDIM + nCol];
      }
    }
    __syncthreads();
    // ---- gate / state update ----
    for (int e = tid; e < BSZ * HDIM; e += 1024) {
      const int b = e / HDIM, j = e % HDIM;
      const float gi = G[b * GDIM + j];
      const float gf = G[b * GDIM + HDIM + j];
      const float gg = G[b * GDIM + 2 * HDIM + j];
      const float go = G[b * GDIM + 3 * HDIM + j];
      float c = Cs[e];
      c = sigm(gf) * c + sigm(gi) * tanhf(gg);
      const float h = sigm(go) * tanhf(c);
      Cs[e] = c;
      Hs[b * HPAD + j] = (_Float16)h;
      hout[((size_t)b * TSEQ + t) * DSTR + dirOff + j] = h;
    }
    __syncthreads();
  }
}

// ---------------------------------------------------------------------------
// emissions[bt,k] = sum_d relu(h[bt,d]) * w_out[k,d] + b_out[k]   (h stride DSTR)
// ---------------------------------------------------------------------------
__global__ void emissions_kernel(const float* __restrict__ h,
                                 const float* __restrict__ wout,
                                 const float* __restrict__ bout,
                                 float* __restrict__ em)
{
  int gid = blockIdx.x * blockDim.x + threadIdx.x;
  if (gid >= MROWS * KTAGS) return;
  int bt = gid / KTAGS, k = gid % KTAGS;
  const float* hr = h + (size_t)bt * DSTR;
  const float* wr = wout + k * (2 * HDIM);
  float s = bout[k];
  for (int d = 0; d < 2 * HDIM; ++d) {
    float v = hr[d];
    v = v > 0.f ? v : 0.f;
    s += v * wr[d];
  }
  em[gid] = s;
}

// ---------------------------------------------------------------------------
// CRF negative log-likelihood. 1 block, 64 threads (one batch element each).
// ---------------------------------------------------------------------------
__global__ __launch_bounds__(64) void crf_kernel(
    const float* __restrict__ em, const unsigned char* __restrict__ mask,
    const int* __restrict__ y, const float* __restrict__ start,
    const float* __restrict__ endv, const float* __restrict__ trans,
    float* __restrict__ out)
{
  __shared__ float sS[KTAGS], sE[KTAGS], sT[KTAGS * KTAGS], red[64];
  const int tid = threadIdx.x;
  if (tid < KTAGS) { sS[tid] = start[tid]; sE[tid] = endv[tid]; }
  if (tid < KTAGS * KTAGS) sT[tid] = trans[tid];
  __syncthreads();

  const int b = tid;
  const float* eb = em + (size_t)b * TSEQ * KTAGS;
  const int* yb = y + b * TSEQ;
  const unsigned char* mb = mask + b * TSEQ;

  // numerator
  float num = sS[yb[0]] + eb[yb[0]];
  int cnt = 1;
  for (int t = 1; t < TSEQ; ++t) {
    if (mb[t]) {
      num += eb[t * KTAGS + yb[t]] + sT[yb[t - 1] * KTAGS + yb[t]];
      ++cnt;
    }
  }
  num += sE[yb[cnt - 1]];

  // forward algorithm (log-space)
  float alpha[KTAGS];
#pragma unroll
  for (int k = 0; k < KTAGS; ++k) alpha[k] = sS[k] + eb[k];
  for (int t = 1; t < TSEQ; ++t) {
    if (!mb[t]) continue;
    float nxt[KTAGS];
#pragma unroll
    for (int k = 0; k < KTAGS; ++k) {
      float mx = -3.4e38f;
#pragma unroll
      for (int j = 0; j < KTAGS; ++j) mx = fmaxf(mx, alpha[j] + sT[j * KTAGS + k]);
      float s = 0.f;
#pragma unroll
      for (int j = 0; j < KTAGS; ++j) s += __expf(alpha[j] + sT[j * KTAGS + k] - mx);
      nxt[k] = mx + __logf(s) + eb[t * KTAGS + k];
    }
#pragma unroll
    for (int k = 0; k < KTAGS; ++k) alpha[k] = nxt[k];
  }
  float mx = -3.4e38f;
#pragma unroll
  for (int k = 0; k < KTAGS; ++k) mx = fmaxf(mx, alpha[k] + sE[k]);
  float s = 0.f;
#pragma unroll
  for (int k = 0; k < KTAGS; ++k) s += __expf(alpha[k] + sE[k] - mx);
  const float den = mx + __logf(s);

  red[tid] = num - den;
  __syncthreads();
  for (int st = 32; st > 0; st >>= 1) {
    if (tid < st) red[tid] += red[tid + st];
    __syncthreads();
  }
  if (tid == 0) out[0] = -red[0];
}

// ---------------------------------------------------------------------------
extern "C" void kernel_launch(void* const* d_in, const int* in_sizes, int n_in,
                              void* d_out, int out_size, void* d_ws, size_t ws_size,
                              hipStream_t stream) {
  (void)in_sizes; (void)n_in; (void)out_size; (void)ws_size;
  const float*         emb    = (const float*)d_in[0];
  const unsigned char* mask   = (const unsigned char*)d_in[1];
  const int*           y      = (const int*)d_in[2];
  const float* w_ih0f = (const float*)d_in[3];
  const float* w_hh0f = (const float*)d_in[4];
  const float* b_ih0f = (const float*)d_in[5];
  const float* b_hh0f = (const float*)d_in[6];
  const float* w_ih0b = (const float*)d_in[7];
  const float* w_hh0b = (const float*)d_in[8];
  const float* b_ih0b = (const float*)d_in[9];
  const float* b_hh0b = (const float*)d_in[10];
  const float* w_ih1f = (const float*)d_in[11];
  const float* w_hh1f = (const float*)d_in[12];
  const float* b_ih1f = (const float*)d_in[13];
  const float* b_hh1f = (const float*)d_in[14];
  const float* w_ih1b = (const float*)d_in[15];
  const float* w_hh1b = (const float*)d_in[16];
  const float* b_ih1b = (const float*)d_in[17];
  const float* b_hh1b = (const float*)d_in[18];
  const float* w_out  = (const float*)d_in[19];
  const float* b_out  = (const float*)d_in[20];
  const float* crf_s  = (const float*)d_in[21];
  const float* crf_e  = (const float*)d_in[22];
  const float* crf_t  = (const float*)d_in[23];
  float* out = (float*)d_out;

  // workspace partition
  char* ws = (char*)d_ws;
  size_t off = 0;
  auto wsalloc = [&](size_t bytes) -> char* {
    char* p = ws + off;
    off += (bytes + 255) & ~(size_t)255;
    return p;
  };
  float*    xpf     = (float*)wsalloc((size_t)MROWS * GDIM * 4);
  float*    xpb     = (float*)wsalloc((size_t)MROWS * GDIM * 4);
  float*    hout0   = (float*)wsalloc((size_t)MROWS * DSTR * 4);
  float*    hout1   = (float*)wsalloc((size_t)MROWS * DSTR * 4);
  _Float16* whh0f16 = (_Float16*)wsalloc((size_t)GDIM * HPAD * 2);
  _Float16* whh0b16 = (_Float16*)wsalloc((size_t)GDIM * HPAD * 2);
  _Float16* whh1f16 = (_Float16*)wsalloc((size_t)GDIM * HPAD * 2);
  _Float16* whh1b16 = (_Float16*)wsalloc((size_t)GDIM * HPAD * 2);
  _Float16* wih0f16 = (_Float16*)wsalloc((size_t)GDIM * EDIM * 2);
  _Float16* wih0b16 = (_Float16*)wsalloc((size_t)GDIM * EDIM * 2);
  _Float16* wih1f16 = (_Float16*)wsalloc((size_t)GDIM * DSTR * 2);
  _Float16* wih1b16 = (_Float16*)wsalloc((size_t)GDIM * DSTR * 2);
  float*    emis    = (float*)wsalloc((size_t)MROWS * KTAGS * 4);

  const size_t lstmLds = (size_t)BSZ * GDIM * 4 + (size_t)BSZ * HPAD * 2 +
                         (size_t)BSZ * HDIM * 4; // 284,672 B < 320 KB/WGP
  (void)hipFuncSetAttribute((const void*)lstm_layer_kernel,
                            hipFuncAttributeMaxDynamicSharedMemorySize,
                            (int)lstmLds);

  // 1) weight conversions to padded f16
  {
    int n = GDIM * HPAD;
    dim3 g((n + 255) / 256);
    convpad_kernel<<<g, 256, 0, stream>>>(w_hh0f, whh0f16, HDIM, HPAD, n);
    convpad_kernel<<<g, 256, 0, stream>>>(w_hh0b, whh0b16, HDIM, HPAD, n);
    convpad_kernel<<<g, 256, 0, stream>>>(w_hh1f, whh1f16, HDIM, HPAD, n);
    convpad_kernel<<<g, 256, 0, stream>>>(w_hh1b, whh1b16, HDIM, HPAD, n);
  }
  {
    int n = GDIM * EDIM;
    dim3 g((n + 255) / 256);
    convpad_kernel<<<g, 256, 0, stream>>>(w_ih0f, wih0f16, EDIM, EDIM, n);
    convpad_kernel<<<g, 256, 0, stream>>>(w_ih0b, wih0b16, EDIM, EDIM, n);
  }
  {
    int n = GDIM * DSTR;
    dim3 g((n + 255) / 256);
    convpad_kernel<<<g, 256, 0, stream>>>(w_ih1f, wih1f16, 2 * HDIM, DSTR, n);
    convpad_kernel<<<g, 256, 0, stream>>>(w_ih1b, wih1b16, 2 * HDIM, DSTR, n);
  }

  // 2) zero hout0 (pad columns must be 0 for layer-1 GEMM)
  {
    int n = MROWS * DSTR;
    zero_kernel<<<(n + 255) / 256, 256, 0, stream>>>(hout0, n);
  }

  // 3) layer 0 input projections (Kpad=1024)
  dim3 gg(MROWS / 32, GDIM / 80);
  gemm_xp_kernel<<<gg, 320, 0, stream>>>(emb, wih0f16, EDIM, b_ih0f, b_hh0f, xpf);
  gemm_xp_kernel<<<gg, 320, 0, stream>>>(emb, wih0b16, EDIM, b_ih0b, b_hh0b, xpb);

  // 4) layer 0 bidirectional scan
  lstm_layer_kernel<<<2, 1024, lstmLds, stream>>>(xpf, xpb, whh0f16, whh0b16, hout0);

  // 5) layer 1 input projections (Kpad=416), reuse xp buffers
  gemm_xp_kernel<<<gg, 320, 0, stream>>>(hout0, wih1f16, DSTR, b_ih1f, b_hh1f, xpf);
  gemm_xp_kernel<<<gg, 320, 0, stream>>>(hout0, wih1b16, DSTR, b_ih1b, b_hh1b, xpb);

  // 6) layer 1 bidirectional scan
  lstm_layer_kernel<<<2, 1024, lstmLds, stream>>>(xpf, xpb, whh1f16, whh1b16, hout1);

  // 7) relu + emissions projection (K=9)
  emissions_kernel<<<(MROWS * KTAGS + 255) / 256, 256, 0, stream>>>(hout1, w_out, b_out, emis);

  // 8) CRF NLL -> scalar
  crf_kernel<<<1, 64, 0, stream>>>(emis, mask, y, crf_s, crf_e, crf_t, out);
}
